// GRAPHX_REFINE_MODEL_21242908246267
// MI455X (gfx1250) — matrix operand order repaired and checked by reading, not compiled, not run on hardware
//
#include <hip/hip_runtime.h>
#include <math.h>
#include <stdint.h>

typedef __attribute__((ext_vector_type(16))) _Float16 v16h;
typedef __attribute__((ext_vector_type(2)))  _Float16 v2h;
typedef __attribute__((ext_vector_type(8)))  float    v8f;

static constexpr int B_ = 8;
static constexpr int N_ = 2048;

// ---------------------------------------------------------------- world2cam
__global__ void world2cam_kernel(const float* __restrict__ xyz,
                                 const float* __restrict__ az,
                                 const float* __restrict__ el,
                                 float* __restrict__ out, int total, int Npts) {
  int t = blockIdx.x * blockDim.x + threadIdx.x;
  if (t >= total) return;
  int b = t / Npts;
  float ca = cosf(az[b]), sa = sinf(az[b]);
  float ce = cosf(el[b]), se = sinf(el[b]);
  const float* p = xyz + (size_t)t * 3;
  float x = p[0], y = p[1], z = p[2];
  float o0 =  ce * ca * x + ce * sa * y + se * z;
  float o1 = -sa * x + ca * y;
  float o2 = -se * ca * x - se * sa * y + ce * z;
  float* q = out + (size_t)t * 3;
  q[0] = o0; q[1] = o1; q[2] = o2 - 2.0f;
}

// ------------------------------------------------------- bilinear projection
__global__ void project_kernel(const float* __restrict__ feat, int C, int H, int W,
                               const float* __restrict__ cam, float* __restrict__ proj,
                               int projStride, int chOff, int Npts) {
  int pt = blockIdx.x;               // b*N + n
  int b = pt / Npts;
  const float* p = cam + (size_t)pt * 3;
  float X = p[0], Y = p[1], Z = fabsf(p[2]);
  float w = fminf(fmaxf(420.f * X / Z + 111.5f, 0.f), 223.f);
  float h = fminf(fmaxf(420.f * Y / Z + 111.5f, 0.f), 223.f);
  float x = w * (float)(W - 1) / 223.f;
  float y = h * (float)(H - 1) / 223.f;
  float x1f = floorf(x), y1f = floorf(y), x2f = ceilf(x), y2f = ceilf(y);
  int i1 = (int)x1f, j1 = (int)y1f, i2 = (int)x2f, j2 = (int)y2f;
  float w11 = (x2f - x) * (y2f - y), w21 = (x - x1f) * (y2f - y);
  float w12 = (x2f - x) * (y - y1f), w22 = (x - x1f) * (y - y1f);
  const float* fb = feat + (size_t)b * C * H * W;
  for (int c = threadIdx.x; c < C; c += blockDim.x) {
    const float* fc = fb + (size_t)c * H * W;
    float q11 = fc[i1 * W + j1];
    float q12 = fc[i1 * W + j2];
    float q21 = fc[i2 * W + j1];
    float q22 = fc[i2 * W + j2];
    proj[(size_t)pt * projStride + chOff + c] = q11 * w11 + q21 * w21 + q12 * w12 + q22 * w22;
  }
}

// ------------------------------------------------- farthest point sampling
__global__ void fps_kernel(const float* __restrict__ xyz, int Nin, int npoint,
                           int* __restrict__ idx_out) {
  int b = blockIdx.x;
  const float* p = xyz + (size_t)b * Nin * 3;
  __shared__ float dist[2048];
  __shared__ float rmax[256];
  __shared__ int   rarg[256];
  __shared__ int   sFar;
  int tid = threadIdx.x;
  for (int n = tid; n < Nin; n += blockDim.x) dist[n] = 1e10f;
  if (tid == 0) sFar = 0;
  __syncthreads();
  for (int i = 0; i < npoint; ++i) {
    int far = sFar;
    if (tid == 0) idx_out[b * npoint + i] = far;
    float cx = p[far * 3 + 0], cy = p[far * 3 + 1], cz = p[far * 3 + 2];
    float bm = -1.f; int ba = 0x7fffffff;
    for (int n = tid; n < Nin; n += blockDim.x) {
      float dx = p[n * 3] - cx, dy = p[n * 3 + 1] - cy, dz = p[n * 3 + 2] - cz;
      float d = dx * dx + dy * dy + dz * dz;
      float nd = fminf(dist[n], d);
      dist[n] = nd;
      if (nd > bm || (nd == bm && n < ba)) { bm = nd; ba = n; }
    }
    rmax[tid] = bm; rarg[tid] = ba;
    __syncthreads();
    for (int s = 128; s > 0; s >>= 1) {
      if (tid < s) {
        if (rmax[tid + s] > rmax[tid] ||
            (rmax[tid + s] == rmax[tid] && rarg[tid + s] < rarg[tid])) {
          rmax[tid] = rmax[tid + s]; rarg[tid] = rarg[tid + s];
        }
      }
      __syncthreads();
    }
    if (tid == 0) sFar = rarg[0];
    __syncthreads();
  }
}

// ------------------------------------------------------------- gather xyz
__global__ void gather3_kernel(const float* __restrict__ xyz, const int* __restrict__ idx,
                               float* __restrict__ out, int Nin, int S, int total) {
  int t = blockIdx.x * blockDim.x + threadIdx.x;
  if (t >= total) return;
  int b = t / S;
  int id = idx[t];
  const float* s = xyz + ((size_t)b * Nin + id) * 3;
  float* o = out + (size_t)t * 3;
  o[0] = s[0]; o[1] = s[1]; o[2] = s[2];
}

// -------------------------------------------------------------- ball query
__global__ void ball_kernel(const float* __restrict__ xyz, const float* __restrict__ nxyz,
                            int Nin, int S, int ns, float radius,
                            int* __restrict__ idx, int total) {
  int t = blockIdx.x * blockDim.x + threadIdx.x;
  if (t >= total) return;
  int b = t / S;
  const float* q = nxyz + (size_t)t * 3;
  const float* p = xyz + (size_t)b * Nin * 3;
  float r2 = radius * radius;
  int* o = idx + (size_t)t * ns;
  int cnt = 0;
  for (int n = 0; n < Nin && cnt < ns; ++n) {
    float dx = p[n * 3] - q[0], dy = p[n * 3 + 1] - q[1], dz = p[n * 3 + 2] - q[2];
    if (dx * dx + dy * dy + dz * dz <= r2) o[cnt++] = n;
  }
  int f = (cnt > 0) ? o[0] : 0;
  for (int k = cnt; k < ns; ++k) o[k] = f;
}

// ---------------------------------------------------------------- grouping
__global__ void group_kernel(const float* __restrict__ xyz, const float* __restrict__ pts,
                             const int* __restrict__ idx, const float* __restrict__ nxyz,
                             int Nin, int S, int ns, int Cin,
                             float* __restrict__ out, size_t total) {
  size_t t = (size_t)blockIdx.x * blockDim.x + threadIdx.x;
  if (t >= total) return;
  size_t g = t / ns;                // b*S + s
  int b = (int)(g / S);
  int id = idx[t];
  const float* sp = xyz + ((size_t)b * Nin + id) * 3;
  const float* nq = nxyz + g * 3;
  int K0 = 3 + Cin;
  float* o = out + t * (size_t)K0;
  o[0] = sp[0] - nq[0]; o[1] = sp[1] - nq[1]; o[2] = sp[2] - nq[2];
  if (pts) {
    const float* pr = pts + ((size_t)b * Nin + id) * Cin;
    for (int c = 0; c < Cin; ++c) o[3 + c] = pr[c];
  }
}

// --------------------------------------------- operand pre-swizzle to f16
// Ah layout: [mrg = Mpad/16][kb = Kpad/32][lane:32][i:16] halves.
// Lane fragment semantics (V_WMMA 16-bit A): lane = h*16+r, half i -> K =
// (i&7) + ((i>>3)<<4) + 8*h, row = mrg*16 + r.
__global__ void convertA_kernel(const float* __restrict__ A, _Float16* __restrict__ Ah,
                                int M, int K, int KB, size_t totalPairs) {
  size_t p = (size_t)blockIdx.x * blockDim.x + threadIdx.x;
  if (p >= totalPairs) return;
  size_t idx = p << 1;
  int i  = (int)(idx & 15);
  size_t rest = idx >> 4;
  int ln = (int)(rest & 31);
  rest >>= 5;
  int kb  = (int)(rest % (size_t)KB);
  int mrg = (int)(rest / (size_t)KB);
  int h = ln >> 4, r = ln & 15;
  int kk = (i & 7) + ((i >> 3) << 4) + h * 8;
  int row = mrg * 16 + r;
  int gk = kb * 32 + kk;
  float f0 = (row < M && gk < K)       ? A[(size_t)row * K + gk]     : 0.f;
  float f1 = (row < M && (gk + 1) < K) ? A[(size_t)row * K + gk + 1] : 0.f;
  v2h pk; pk[0] = (_Float16)f0; pk[1] = (_Float16)f1;
  *(v2h*)&Ah[idx] = pk;
}

// Bh layout: [kb = Kpad/32][cg = Npad/16][lane:32][i:16] halves.
// Lane fragment semantics (V_WMMA 16-bit B): lane = h*16 + col, half i -> K =
// (i&7) + ((i>>3)<<4) + 8*h.
__global__ void convertB_kernel(const float* __restrict__ W, _Float16* __restrict__ Bh,
                                int K, int N, int NG, size_t totalPairs) {
  size_t p = (size_t)blockIdx.x * blockDim.x + threadIdx.x;
  if (p >= totalPairs) return;
  size_t idx = p << 1;
  int i  = (int)(idx & 15);
  size_t rest = idx >> 4;
  int ln = (int)(rest & 31);
  rest >>= 5;
  int cg = (int)(rest % (size_t)NG);
  int kb = (int)(rest / (size_t)NG);
  int khigh = ln >> 4, cc = ln & 15;
  int kk = (i & 7) + ((i >> 3) << 4) + khigh * 8;
  int gk = kb * 32 + kk;
  int col = cg * 16 + cc;
  float f0 = (gk < K && col < N)       ? W[(size_t)gk * N + col]       : 0.f;
  float f1 = ((gk + 1) < K && col < N) ? W[(size_t)(gk + 1) * N + col] : 0.f;
  v2h pk; pk[0] = (_Float16)f0; pk[1] = (_Float16)f1;
  *(v2h*)&Bh[idx] = pk;
}

// ------------------------------------------------ WMMA GEMM: C = A*W + bias
// Operands pre-swizzled f16 (Ah, Bh). Block tile 64x64, 128 threads = 4
// wave32s, each wave a 32x32 tile (4 accumulators). 64 K per step => 8
// v_wmma per body. A fragments: direct global v16h (global_load_b128 x2).
// B panels (reused across all M-blocks): double-buffered async copy into LDS
// (global_load_async_to_lds_b128, ASYNCcnt + barrier sync), then ds_load_b128.
__global__ void gemm_bias_wmma(const _Float16* __restrict__ Ah,
                               const _Float16* __restrict__ Bh,
                               const float* __restrict__ bias, float* __restrict__ C,
                               int M, int N, int KB, int NG) {
  // sB: [buf:2][j:2][cg:4][lane:32][i:16] halves = 16 KB
  __shared__ __align__(16) _Float16 sB[8192];
  const int tid  = threadIdx.x;
  const int lane = tid & 31;
  const int wave = tid >> 5;          // 0..3
  const int wm = wave >> 1, wn = wave & 1;
  const int tM = blockIdx.y * 64, tN = blockIdx.x * 64;
  const int cgBase = blockIdx.x * 4;
  const int mrgBase = blockIdx.y * 4 + wm * 2;
  v8f acc00 = {}, acc01 = {}, acc10 = {}, acc11 = {};

  // Stage the 2-kb (64-K) B panel for step kb2s into LDS buffer `buf`.
  auto stage = [&](int buf, int kb2s) {
    for (int c = tid; c < 256; c += 128) {        // 256 x 16B chunks per kb
#pragma unroll
      for (int j = 0; j < 2; ++j) {
        const _Float16* gp = Bh + ((size_t)(kb2s + j) * NG + cgBase) * 512 + (size_t)c * 8;
        unsigned lp = (unsigned)(uintptr_t)&sB[(buf * 2 + j) * 2048 + c * 8];
        unsigned long long ga = (unsigned long long)(uintptr_t)gp;
        asm volatile("global_load_async_to_lds_b128 %0, %1, off"
                     :: "v"(lp), "v"(ga) : "memory");
      }
    }
  };

  stage(0, 0);
  for (int kb2 = 0; kb2 < KB; kb2 += 2) {
    int buf = (kb2 >> 1) & 1;
    bool more = (kb2 + 2) < KB;
    if (more) stage(buf ^ 1, kb2 + 2);
    if (more) {
      asm volatile("s_wait_asynccnt 4" ::: "memory");   // this buf's 4 copies done
    } else {
      asm volatile("s_wait_asynccnt 0" ::: "memory");
    }
    __syncthreads();                                    // all waves staged

    const v16h* pA = (const v16h*)Ah;
    const v16h* pB = (const v16h*)sB;
    if (more) {
      __builtin_prefetch(&Ah[((size_t)mrgBase * KB + kb2 + 2) * 512], 0, 1);
    }
#pragma unroll
    for (int j = 0; j < 2; ++j) {
      int kb = kb2 + j;
      v16h a0 = pA[((size_t)(mrgBase + 0) * KB + kb) * 32 + lane];
      v16h a1 = pA[((size_t)(mrgBase + 1) * KB + kb) * 32 + lane];
      v16h b0 = pB[((buf * 2 + j) * 4 + wn * 2 + 0) * 32 + lane];
      v16h b1 = pB[((buf * 2 + j) * 4 + wn * 2 + 1) * 32 + lane];
      acc00 = __builtin_amdgcn_wmma_f32_16x16x32_f16(false, a0, false, b0, (short)0, acc00, false, false);
      acc01 = __builtin_amdgcn_wmma_f32_16x16x32_f16(false, a0, false, b1, (short)0, acc01, false, false);
      acc10 = __builtin_amdgcn_wmma_f32_16x16x32_f16(false, a1, false, b0, (short)0, acc10, false, false);
      acc11 = __builtin_amdgcn_wmma_f32_16x16x32_f16(false, a1, false, b1, (short)0, acc11, false, false);
    }
    __syncthreads();                                    // done reading this buf
  }

  // C/D layout: VGPR r -> M=r (lanes 0-15) / M=r+8 (lanes 16-31), N=lane&15
  const int rofs = (lane >= 16) ? 8 : 0;
  const int cl = lane & 15;
#pragma unroll
  for (int mi = 0; mi < 2; ++mi) {
#pragma unroll
    for (int ni = 0; ni < 2; ++ni) {
      const v8f& a = (mi == 0 ? (ni == 0 ? acc00 : acc01)
                              : (ni == 0 ? acc10 : acc11));
      int col = tN + wn * 32 + ni * 16 + cl;
      if (col >= N) continue;
      float bv = bias ? bias[col] : 0.f;
      int rbase = tM + wm * 32 + mi * 16 + rofs;
#pragma unroll
      for (int r = 0; r < 8; ++r) {
        int row = rbase + r;
        if (row < M) C[(size_t)row * N + col] = a[r] + bv;
      }
    }
  }
}

// ----------------------------------------------------------- batchnorm stats
__global__ void bn_stats_kernel(const float* __restrict__ Y, int M, int N,
                                float* __restrict__ stats) {
  int c = blockIdx.x;
  __shared__ float ss[256], ss2[256];
  float s = 0.f, s2 = 0.f;
  for (int r = threadIdx.x; r < M; r += blockDim.x) {
    float v = Y[(size_t)r * N + c];
    s += v; s2 += v * v;
  }
  ss[threadIdx.x] = s; ss2[threadIdx.x] = s2;
  __syncthreads();
  for (int st = 128; st > 0; st >>= 1) {
    if (threadIdx.x < st) {
      ss[threadIdx.x]  += ss[threadIdx.x + st];
      ss2[threadIdx.x] += ss2[threadIdx.x + st];
    }
    __syncthreads();
  }
  if (threadIdx.x == 0) {
    float m = ss[0] / (float)M;
    float var = ss2[0] / (float)M - m * m;
    stats[2 * c] = m;
    stats[2 * c + 1] = rsqrtf(var + 1e-5f);
  }
}

__global__ void bn_relu_kernel(float* __restrict__ Y, const float* __restrict__ stats,
                               const float* __restrict__ g, const float* __restrict__ beta,
                               size_t total, int N) {
  size_t t = (size_t)blockIdx.x * blockDim.x + threadIdx.x;
  if (t >= total) return;
  int c = (int)(t % (size_t)N);
  float v = (Y[t] - stats[2 * c]) * stats[2 * c + 1] * g[c] + beta[c];
  Y[t] = v > 0.f ? v : 0.f;
}

// -------------------------------------------------------------- max pooling
__global__ void maxpool_kernel(const float* __restrict__ in, int groups, int ns, int C,
                               float* __restrict__ out, size_t total) {
  size_t t = (size_t)blockIdx.x * blockDim.x + threadIdx.x;
  if (t >= total) return;
  int c = (int)(t % (size_t)C);
  size_t g = t / (size_t)C;
  const float* p = in + g * (size_t)ns * C + c;
  float m = p[0];
  for (int k = 1; k < ns; ++k) m = fmaxf(m, p[(size_t)k * C]);
  out[t] = m;
}

// --------------------------------------------------------------- concat copy
__global__ void copy_cols_kernel(float* __restrict__ dst, int dstStride, int dstOff,
                                 const float* __restrict__ src, int srcStride, int srcOff,
                                 size_t rows, int cols) {
  size_t t = (size_t)blockIdx.x * blockDim.x + threadIdx.x;
  size_t total = rows * (size_t)cols;
  if (t >= total) return;
  size_t r = t / (size_t)cols;
  int c = (int)(t % (size_t)cols);
  dst[r * dstStride + dstOff + c] = src[r * srcStride + srcOff + c];
}

// ------------------------------------------------- S==1 broadcast "interp"
__global__ void broadcast_kernel(const float* __restrict__ src /*B x C*/,
                                 float* __restrict__ out, int N1, int C, size_t total) {
  size_t t = (size_t)blockIdx.x * blockDim.x + threadIdx.x;
  if (t >= total) return;
  int c = (int)(t % (size_t)C);
  size_t bn = t / (size_t)C;
  int b = (int)(bn / (size_t)N1);
  out[t] = src[(size_t)b * C + c];
}

// -------------------------------------------------------- 3-NN interpolation
__global__ void interp3_kernel(const float* __restrict__ xyz1, const float* __restrict__ xyz2,
                               const float* __restrict__ pts2, int N1, int S, int C,
                               float* __restrict__ out) {
  int t = blockIdx.x;               // b*N1 + n
  int b = t / N1;
  __shared__ int   si[3];
  __shared__ float sw[3];
  if (threadIdx.x == 0) {
    const float* q = xyz1 + (size_t)t * 3;
    float qx = q[0], qy = q[1], qz = q[2];
    float d0 = 1e30f, d1 = 1e30f, d2 = 1e30f;
    int i0 = 0, i1 = 0, i2 = 0;
    const float* base = xyz2 + (size_t)b * S * 3;
    for (int s = 0; s < S; ++s) {
      float dx = base[s * 3] - qx, dy = base[s * 3 + 1] - qy, dz = base[s * 3 + 2] - qz;
      float d = dx * dx + dy * dy + dz * dz;
      if (d < d0)      { d2 = d1; i2 = i1; d1 = d0; i1 = i0; d0 = d; i0 = s; }
      else if (d < d1) { d2 = d1; i2 = i1; d1 = d;  i1 = s; }
      else if (d < d2) { d2 = d;  i2 = s; }
    }
    float r0 = 1.f / (d0 + 1e-8f), r1 = 1.f / (d1 + 1e-8f), r2 = 1.f / (d2 + 1e-8f);
    float sum = r0 + r1 + r2;
    si[0] = i0; si[1] = i1; si[2] = i2;
    sw[0] = r0 / sum; sw[1] = r1 / sum; sw[2] = r2 / sum;
  }
  __syncthreads();
  const float* pb = pts2 + (size_t)b * S * C;
  for (int c = threadIdx.x; c < C; c += blockDim.x) {
    out[(size_t)t * C + c] = pb[si[0] * C + c] * sw[0] +
                             pb[si[1] * C + c] * sw[1] +
                             pb[si[2] * C + c] * sw[2];
  }
}

// ------------------------------------ final 64->3 + sigmoid + residual add
__global__ void disp_final_kernel(const float* __restrict__ x, const float* __restrict__ w,
                                  const float* __restrict__ bias, const float* __restrict__ xyz,
                                  float* __restrict__ out, int total) {
  int t = blockIdx.x * blockDim.x + threadIdx.x;
  if (t >= total) return;
  const float* xr = x + (size_t)t * 64;
  const float* q = xyz + (size_t)t * 3;
  float* o = out + (size_t)t * 3;
  for (int j = 0; j < 3; ++j) {
    float acc = bias[j];
    for (int k = 0; k < 64; ++k) acc += xr[k] * w[k * 3 + j];
    float sg = 1.f / (1.f + expf(-acc));
    o[j] = q[j] + sg * 0.4f - 0.2f;
  }
}

// =========================================================== host orchestration
extern "C" void kernel_launch(void* const* d_in, const int* in_sizes, int n_in,
                              void* d_out, int out_size, void* d_ws, size_t ws_size,
                              hipStream_t stream) {
  (void)in_sizes; (void)n_in; (void)out_size;

  const float* imgf[4] = {(const float*)d_in[0], (const float*)d_in[1],
                          (const float*)d_in[2], (const float*)d_in[3]};
  const float* xyz    = (const float*)d_in[4];
  const float* az     = (const float*)d_in[5];
  const float* el     = (const float*)d_in[6];
  const float* noises = (const float*)d_in[7];

  struct LayerP { const float *w, *b, *g, *beta; };
  int pi = 8;
  auto take = [&](LayerP* dst, int n) {
    for (int i = 0; i < n; ++i) {
      dst[i].w    = (const float*)d_in[pi + 0];
      dst[i].b    = (const float*)d_in[pi + 1];
      dst[i].g    = (const float*)d_in[pi + 2];
      dst[i].beta = (const float*)d_in[pi + 3];
      pi += 4;
    }
  };
  LayerP sa1[3], sa2[3], sa3[3], sa4[3], fp4L[2], fp3L[2], fp2L[2], fp1L[3], dispL[5];
  take(sa1, 3); take(sa2, 3); take(sa3, 3); take(sa4, 3);
  take(fp4L, 2); take(fp3L, 2); take(fp2L, 2); take(fp1L, 3); take(dispL, 5);
  const float* dow = (const float*)d_in[pi + 0];
  const float* dob = (const float*)d_in[pi + 1];

  // Bump allocators over d_ws: persistent grows up, per-stage scratch grows down.
  char* base = (char*)d_ws;
  size_t plo = 0;
  size_t shi = ws_size & ~(size_t)255;
  const size_t scratch_top = shi;
  auto pallocf = [&](size_t n) -> float* {
    float* r = (float*)(base + plo);
    plo += ((n * sizeof(float) + 255) & ~(size_t)255);
    return r;
  };
  auto palloch = [&](size_t n) -> _Float16* {
    _Float16* r = (_Float16*)(base + plo);
    plo += ((n * sizeof(_Float16) + 255) & ~(size_t)255);
    return r;
  };
  auto sallocf = [&](size_t n) -> float* {
    shi -= ((n * sizeof(float) + 255) & ~(size_t)255);
    return (float*)(base + shi);
  };
  auto salloci = [&](size_t n) -> int* {
    shi -= ((n * sizeof(int) + 255) & ~(size_t)255);
    return (int*)(base + shi);
  };
  auto sreset = [&]() { shi = scratch_top; };

  // Reusable pre-swizzled operand buffers (stream-ordered reuse is safe).
  _Float16* AhBuf = palloch((size_t)40 << 20);   // 80 MB
  _Float16* BhBuf = palloch((size_t)2  << 20);   //  4 MB

  auto gemm = [&](const float* A, const float* W, const float* bias, float* Cc,
                  int M, int K, int Nn) {
    int gy = (M + 63) / 64, gx = (Nn + 63) / 64;
    int KB = ((K + 63) & ~63) / 32;   // even # of 32-K blocks
    int NG = gx * 4;
    int MRG = gy * 4;
    size_t pa = (size_t)MRG * KB * 256;
    convertA_kernel<<<(unsigned)((pa + 255) / 256), 256, 0, stream>>>(A, AhBuf, M, K, KB, pa);
    size_t pb = (size_t)KB * NG * 256;
    convertB_kernel<<<(unsigned)((pb + 255) / 256), 256, 0, stream>>>(W, BhBuf, K, Nn, NG, pb);
    dim3 grid(gx, gy);
    gemm_bias_wmma<<<grid, 128, 0, stream>>>(AhBuf, BhBuf, bias, Cc, M, Nn, KB, NG);
  };
  auto bnrelu = [&](float* Y, size_t M, int Nn, const LayerP& L) {
    float* stats = sallocf(2 * (size_t)Nn);
    bn_stats_kernel<<<Nn, 256, 0, stream>>>(Y, (int)M, Nn, stats);
    size_t tot = M * (size_t)Nn;
    bn_relu_kernel<<<(unsigned)((tot + 255) / 256), 256, 0, stream>>>(Y, stats, L.g, L.beta, tot, Nn);
  };
  auto cbr_chain = [&](const float* x0, size_t M, const int* ch, int nl,
                       const LayerP* Ls) -> float* {
    const float* cur = x0;
    float* y = nullptr;
    for (int l = 0; l < nl; ++l) {
      y = sallocf(M * (size_t)ch[l + 1]);
      gemm(cur, Ls[l].w, Ls[l].b, y, (int)M, ch[l], ch[l + 1]);
      bnrelu(y, M, ch[l + 1], Ls[l]);
      cur = y;
    }
    return y;
  };

  auto run_sa = [&](const float* xin, const float* pin, int Nin, int Cin,
                    int npoint, float radius, int ns, const int* ch, const LayerP* Ls,
                    float** oxyz, float** ofeat) {
    sreset();
    int* fidx = salloci((size_t)B_ * npoint);
    fps_kernel<<<B_, 256, 0, stream>>>(xin, Nin, npoint, fidx);
    float* nx = pallocf((size_t)B_ * npoint * 3);
    int totS = B_ * npoint;
    gather3_kernel<<<(totS + 255) / 256, 256, 0, stream>>>(xin, fidx, nx, Nin, npoint, totS);
    int* bidx = salloci((size_t)totS * ns);
    ball_kernel<<<(totS + 255) / 256, 256, 0, stream>>>(xin, nx, Nin, npoint, ns, radius, bidx, totS);
    size_t M = (size_t)totS * ns;
    int K0 = 3 + Cin;
    float* g = sallocf(M * (size_t)K0);
    group_kernel<<<(unsigned)((M + 255) / 256), 256, 0, stream>>>(xin, pin, bidx, nx, Nin, npoint, ns, Cin, g, M);
    float* y = cbr_chain(g, M, ch, 3, Ls);
    int Cout = ch[3];
    float* feat = pallocf((size_t)totS * Cout);
    size_t tot = (size_t)totS * Cout;
    maxpool_kernel<<<(unsigned)((tot + 255) / 256), 256, 0, stream>>>(y, totS, ns, Cout, feat, tot);
    *oxyz = nx; *ofeat = feat;
  };

  auto run_fp = [&](const float* x1, const float* x2, const float* p1, const float* p2,
                    int N1, int S, int C1, int C2, const int* ch, int nl,
                    const LayerP* Ls) -> float* {
    sreset();
    size_t Mm = (size_t)B_ * N1;
    float* interp = sallocf(Mm * (size_t)C2);
    if (S == 1) {
      size_t tot = Mm * (size_t)C2;
      broadcast_kernel<<<(unsigned)((tot + 255) / 256), 256, 0, stream>>>(p2, interp, N1, C2, tot);
    } else {
      interp3_kernel<<<(unsigned)Mm, 128, 0, stream>>>(x1, x2, p2, N1, S, C2, interp);
    }
    const float* xin;
    if (p1) {
      int K0 = C1 + C2;
      float* xc = sallocf(Mm * (size_t)K0);
      size_t t1 = Mm * (size_t)C1;
      copy_cols_kernel<<<(unsigned)((t1 + 255) / 256), 256, 0, stream>>>(xc, K0, 0, p1, C1, 0, Mm, C1);
      size_t t2 = Mm * (size_t)C2;
      copy_cols_kernel<<<(unsigned)((t2 + 255) / 256), 256, 0, stream>>>(xc, K0, C1, interp, C2, 0, Mm, C2);
      xin = xc;
    } else {
      xin = interp;
    }
    float* y = cbr_chain(xin, Mm, ch, nl, Ls);
    int Cout = ch[nl];
    float* out = pallocf(Mm * (size_t)Cout);
    size_t tot = Mm * (size_t)Cout;
    copy_cols_kernel<<<(unsigned)((tot + 255) / 256), 256, 0, stream>>>(out, Cout, 0, y, Cout, 0, Mm, Cout);
    return out;
  };

  // -------- world2cam + projection
  int totPts = B_ * N_;
  float* cam = pallocf((size_t)totPts * 3);
  world2cam_kernel<<<(totPts + 255) / 256, 256, 0, stream>>>(xyz, az, el, cam, totPts, N_);
  float* proj = pallocf((size_t)totPts * 960);
  const int pc[4] = {64, 128, 256, 512};
  const int ph[4] = {112, 56, 28, 14};
  int choff = 0;
  for (int i = 0; i < 4; ++i) {
    project_kernel<<<totPts, 128, 0, stream>>>(imgf[i], pc[i], ph[i], ph[i], cam, proj, 960, choff, N_);
    choff += pc[i];
  }

  // -------- PointNet++ set abstraction
  float *l1x, *l1p, *l2x, *l2p, *l3x, *l3p, *l4p;
  const int ch1[4] = {3, 64, 64, 128};
  run_sa(cam, nullptr, N_, 0, 1024, 0.1f, 64, ch1, sa1, &l1x, &l1p);
  const int ch2[4] = {131, 128, 128, 256};
  run_sa(l1x, l1p, 1024, 128, 384, 0.2f, 64, ch2, sa2, &l2x, &l2p);
  const int ch3[4] = {259, 256, 256, 512};
  run_sa(l2x, l2p, 384, 256, 128, 0.4f, 64, ch3, sa3, &l3x, &l3p);

  // -------- sa4: group-all
  sreset();
  {
    size_t Mm = (size_t)B_ * 128;
    float* x4 = sallocf(Mm * 515);
    size_t t1 = Mm * 3;
    copy_cols_kernel<<<(unsigned)((t1 + 255) / 256), 256, 0, stream>>>(x4, 515, 0, l3x, 3, 0, Mm, 3);
    size_t t2 = Mm * 512;
    copy_cols_kernel<<<(unsigned)((t2 + 255) / 256), 256, 0, stream>>>(x4, 515, 3, l3p, 512, 0, Mm, 512);
    const int ch4[4] = {515, 512, 512, 1024};
    float* y4 = cbr_chain(x4, Mm, ch4, 3, sa4);
    l4p = pallocf((size_t)B_ * 1024);
    size_t tot = (size_t)B_ * 1024;
    maxpool_kernel<<<(unsigned)((tot + 255) / 256), 256, 0, stream>>>(y4, B_, 128, 1024, l4p, tot);
  }

  // -------- feature propagation
  const int chf4[3] = {1536, 512, 512};
  float* nl3p = run_fp(l3x, nullptr, l3p, l4p, 128, 1, 512, 1024, chf4, 2, fp4L);
  const int chf3[3] = {768, 512, 256};
  float* nl2p = run_fp(l2x, l3x, l2p, nl3p, 384, 128, 256, 512, chf3, 2, fp3L);
  const int chf2[3] = {384, 256, 128};
  float* nl1p = run_fp(l1x, l2x, l1p, nl2p, 1024, 384, 128, 256, chf2, 2, fp2L);
  const int chf1[4] = {128, 128, 128, 128};
  float* l0p = run_fp(cam, l1x, nullptr, nl1p, N_, 1024, 0, 128, chf1, 3, fp1L);

  // -------- displacement head
  sreset();
  {
    size_t Mm = (size_t)totPts;
    float* xd = sallocf(Mm * 1120);
    size_t t1 = Mm * 128;
    copy_cols_kernel<<<(unsigned)((t1 + 255) / 256), 256, 0, stream>>>(xd, 1120, 0, l0p, 128, 0, Mm, 128);
    size_t t2 = Mm * 960;
    copy_cols_kernel<<<(unsigned)((t2 + 255) / 256), 256, 0, stream>>>(xd, 1120, 128, proj, 960, 0, Mm, 960);
    size_t t3 = Mm * 32;
    copy_cols_kernel<<<(unsigned)((t3 + 255) / 256), 256, 0, stream>>>(xd, 1120, 1088, noises, 32, 0, Mm, 32);
    const int chd[6] = {1120, 960, 512, 256, 128, 64};
    float* yd = cbr_chain(xd, Mm, chd, 5, dispL);
    disp_final_kernel<<<(totPts + 255) / 256, 256, 0, stream>>>(yd, dow, dob, xyz, (float*)d_out, totPts);
  }
}
